// ResidualBlock_67044439490733
// MI455X (gfx1250) — matrix operand order repaired and checked by reading, not compiled, not run on hardware
//
#include <hip/hip_runtime.h>
#include <math.h>

// ---------------------------------------------------------------------------
// ResidualBlock forward for MI455X (gfx1250), wave32 + WMMA bf16.
// Dominant GEMM: [800x6400]@[6400x6400] (65.5 GFLOP); sa_W (164MB) fits L2.
// All dense ops go through one generic WMMA GEMM (bf16 in, fp32 accum).
// Fragments are fetched from LDS with wide ds_load_b128; edge handling is
// branch-free (clamped loads + selects).
// ---------------------------------------------------------------------------

typedef __bf16 bf16_t;
typedef __attribute__((ext_vector_type(16))) __bf16 v16bf;
typedef __attribute__((ext_vector_type(8)))  __bf16 v8bf;
typedef __attribute__((ext_vector_type(8)))  float  v8f;

#define ACT_NONE 0
#define ACT_RELU 1
#define ACT_SILU 2
#define ACT_GELU 3

// problem sizes
#define Bn   8
#define Cn   64
#define Ln   100
#define Nn   100
#define LNn  10000      // L*N
#define OUTCn 6400      // C*L
#define SIDEn 144
#define DEMBn 128
#define Gn   64
#define TOPKn 20

__device__ __forceinline__ float sigm(float x) { return 1.f / (1.f + __expf(-x)); }

// pack two fp32 -> two bf16 in one dword (low = first)
__device__ __forceinline__ unsigned pack2(float a, float b) {
  unsigned short ua = __builtin_bit_cast(unsigned short, (bf16_t)a);
  unsigned short ub = __builtin_bit_cast(unsigned short, (bf16_t)b);
  return (unsigned)ua | ((unsigned)ub << 16);
}

// ---------------------------------------------------------------------------
// Generic batched GEMM:  C[m,n] = act( sum_k A[m,k]*B(k,n) + bias[n] + Add[m,n] )
//   A: row-major M x K.  B(k,n) = transB ? Bw[n*ldb+k] : Bw[k*ldb+n].
// Block = 128 threads (4 waves) -> one 64x64 C tile; wave w owns rows 16w..16w+15
// with 4 f32 accumulators across the 64 columns.
// LDS per k-step: A tile 64x32 bf16 [row][k]; B tile transposed 64x32 bf16 [n][k]
// (96B pitch, 32B-aligned rows) so:
//   A fragment = 2x ds_load_b128 (K runs 16hf..+7 and 16hf+8..+15 of own row)
//   B fragment = 1x 32B load     (K run 16hf..+15 of own column row)
// k = 16hf+s for both A and B (contiguous order).  All lda/ldb are multiples of 4 ->
// fp32 staging uses float4 global loads; bf16 pairs packed, b128/b64 DS stores.
// Out-of-range staging uses clamped addresses + selects (no exec branching).
// ---------------------------------------------------------------------------
__global__ void __launch_bounds__(128)
gemm_wmma(const float* __restrict__ A, long long lda, long long strideA,
          const float* __restrict__ Bw, long long ldb, long long strideB, int transB,
          float* __restrict__ Cc, long long ldc, long long strideC,
          const float* __restrict__ bias,
          const float* __restrict__ Add,
          int M, int N, int K, int act)
{
  __shared__ __align__(32) unsigned As_u[64][24];   // 64 rows x 32 bf16 (+pad)
  __shared__ __align__(32) unsigned BsT_u[64][24];  // 64 cols x 32 bf16 (+pad)

  const int lane = threadIdx.x & 31;
  const int wave = threadIdx.x >> 5;
  const int hf   = lane >> 4;
  const int l16  = lane & 15;
  const long long n0 = (long long)blockIdx.x * 64;
  const long long mBase = (long long)blockIdx.y * 64;
  const long long m0 = mBase + wave * 16;
  const int z = blockIdx.z;
  A  += (long long)z * strideA;
  Bw += (long long)z * strideB;
  Cc += (long long)z * strideC;
  const float* Addp = Add ? (Add + (long long)z * strideC) : (const float*)0;

  v8f acc[4];
#pragma unroll
  for (int t = 0; t < 4; ++t) acc[t] = (v8f){0.f,0.f,0.f,0.f,0.f,0.f,0.f,0.f};

  for (int k0 = 0; k0 < K; k0 += 32) {
    // ---------------- stage A tile 64x32 -> As_u ----------------
    {
      int row = threadIdx.x >> 1;
      int kc  = (threadIdx.x & 1) * 16;
      long long m = mBase + row;
      int kg0 = k0 + kc;
      float tv[16];
      if (m < (long long)M && kg0 + 16 <= K) {       // fast interior path
        const float4* p = (const float4*)(A + m * lda + kg0);
        float4 q0 = p[0], q1 = p[1], q2 = p[2], q3 = p[3];
        tv[0]=q0.x; tv[1]=q0.y; tv[2]=q0.z; tv[3]=q0.w;
        tv[4]=q1.x; tv[5]=q1.y; tv[6]=q1.z; tv[7]=q1.w;
        tv[8]=q2.x; tv[9]=q2.y; tv[10]=q2.z; tv[11]=q2.w;
        tv[12]=q3.x; tv[13]=q3.y; tv[14]=q3.z; tv[15]=q3.w;
        if (k0 + 32 < K) __builtin_prefetch(A + m * lda + kg0 + 32, 0, 1);
      } else {                                        // branch-free edge path
        long long mc = (m < (long long)M) ? m : (long long)(M - 1);
        const float* Arow = A + mc * lda;
#pragma unroll
        for (int j = 0; j < 16; ++j) {
          int kg = kg0 + j;
          int kcl = (kg < K) ? kg : (K - 1);
          float v = Arow[kcl];                        // always in-bounds
          tv[j] = (m < (long long)M && kg < K) ? v : 0.f;
        }
      }
      unsigned u[8];
#pragma unroll
      for (int i = 0; i < 8; ++i) u[i] = pack2(tv[2*i], tv[2*i+1]);
      uint4* d = (uint4*)&As_u[row][kc >> 1];
      uint4 w0; w0.x=u[0]; w0.y=u[1]; w0.z=u[2]; w0.w=u[3];
      uint4 w1; w1.x=u[4]; w1.y=u[5]; w1.z=u[6]; w1.w=u[7];
      d[0] = w0; d[1] = w1;
    }

    // ---------------- stage B tile (transposed) -> BsT_u ----------------
    if (transB) {
      // B is [N][K] row-major: contiguous in k, same orientation as BsT.
      int n  = threadIdx.x >> 1;
      int kc = (threadIdx.x & 1) * 16;
      int kg0 = k0 + kc;
      float tv[16];
      if (kg0 + 16 <= K) {
        const float4* p = (const float4*)(Bw + (n0 + n) * ldb + kg0);
        float4 q0 = p[0], q1 = p[1], q2 = p[2], q3 = p[3];
        tv[0]=q0.x; tv[1]=q0.y; tv[2]=q0.z; tv[3]=q0.w;
        tv[4]=q1.x; tv[5]=q1.y; tv[6]=q1.z; tv[7]=q1.w;
        tv[8]=q2.x; tv[9]=q2.y; tv[10]=q2.z; tv[11]=q2.w;
        tv[12]=q3.x; tv[13]=q3.y; tv[14]=q3.z; tv[15]=q3.w;
      } else {
        const float* Brow = Bw + (n0 + n) * ldb;
#pragma unroll
        for (int j = 0; j < 16; ++j) {
          int kg = kg0 + j;
          int kcl = (kg < K) ? kg : (K - 1);
          float v = Brow[kcl];
          tv[j] = (kg < K) ? v : 0.f;
        }
      }
      unsigned u[8];
#pragma unroll
      for (int i = 0; i < 8; ++i) u[i] = pack2(tv[2*i], tv[2*i+1]);
      uint4* d = (uint4*)&BsT_u[n][kc >> 1];
      uint4 w0; w0.x=u[0]; w0.y=u[1]; w0.z=u[2]; w0.w=u[3];
      uint4 w1; w1.x=u[4]; w1.y=u[5]; w1.z=u[6]; w1.w=u[7];
      d[0] = w0; d[1] = w1;
    } else {
      // B is [K][N] row-major: read 4 coalesced float4 row segments,
      // transpose into 4 packed b64 column writes (clamped rows, selects).
      int nq = threadIdx.x & 15;   // n quad
      int kq = threadIdx.x >> 4;   // k quad (0..7)
      int n = nq * 4, k = kq * 4;
      float4 r[4];
#pragma unroll
      for (int rr = 0; rr < 4; ++rr) {
        int kg = k0 + k + rr;
        int kcl = (kg < K) ? kg : (K - 1);
        float4 q = *(const float4*)(Bw + (long long)kcl * ldb + n0 + n);
        bool ok = (kg < K);
        r[rr].x = ok ? q.x : 0.f; r[rr].y = ok ? q.y : 0.f;
        r[rr].z = ok ? q.z : 0.f; r[rr].w = ok ? q.w : 0.f;
      }
      const float* f0 = (const float*)&r[0];
      const float* f1 = (const float*)&r[1];
      const float* f2 = (const float*)&r[2];
      const float* f3 = (const float*)&r[3];
#pragma unroll
      for (int c = 0; c < 4; ++c) {
        uint2 w; w.x = pack2(f0[c], f1[c]); w.y = pack2(f2[c], f3[c]);
        *(uint2*)&BsT_u[n + c][k >> 1] = w;
      }
    }
    __syncthreads();

    // ------- load A fragment + ALL B fragments, then chain 4 WMMAs -------
    const int rowA = wave * 16 + l16;
    v8bf a0 = *(const v8bf*)&As_u[rowA][8 * hf];       // K = 16hf .. 16hf+7
    v8bf a1 = *(const v8bf*)&As_u[rowA][8 * hf + 4];   // K = 16hf+8 .. +7
    v16bf afrag = __builtin_shufflevector(a0, a1,
        0,1,2,3,4,5,6,7,8,9,10,11,12,13,14,15);

    v16bf bfr[4];
#pragma unroll
    for (int t = 0; t < 4; ++t)
      bfr[t] = *(const v16bf*)&BsT_u[t * 16 + l16][8 * hf];  // K = 16hf..+15

#pragma unroll
    for (int t = 0; t < 4; ++t)
      acc[t] = __builtin_amdgcn_wmma_f32_16x16x32_bf16(
          false, afrag, false, bfr[t], (short)0, acc[t], false, false);
    __syncthreads();
  }

  // ---- epilogue: D layout (VGPR r -> M = r + 8*hf, N = l16) ----
#pragma unroll
  for (int t = 0; t < 4; ++t) {
    long long n = n0 + t * 16 + l16;
#pragma unroll
    for (int r = 0; r < 8; ++r) {
      long long m = m0 + 8 * hf + r;
      if (m < (long long)M) {
        float v = acc[t][r];
        if (bias) v += bias[n];
        if (Addp) v += Addp[m * ldc + n];
        if (act == ACT_RELU)      v = fmaxf(v, 0.f);
        else if (act == ACT_SILU) v = v * sigm(v);
        else if (act == ACT_GELU) v = 0.5f * v * (1.f + erff(v * 0.70710678118654752f));
        Cc[m * ldc + n] = v;
      }
    }
  }
}

// ---------------------------------------------------------------------------
// Small / elementwise kernels
// ---------------------------------------------------------------------------

// ci [B][SIDE][LN] -> ci_t [(b*LN+t)][SIDE]   (coalesced reads)
__global__ void k_transpose_ci(const float* __restrict__ ci, float* __restrict__ ci_t)
{
  long long idx = (long long)blockIdx.x * blockDim.x + threadIdx.x;
  if (idx >= (long long)Bn * SIDEn * LNn) return;
  int t = (int)(idx % LNn);
  int s = (int)((idx / LNn) % SIDEn);
  int b = (int)(idx / ((long long)SIDEn * LNn));
  ci_t[((long long)b * LNn + t) * SIDEn + s] = ci[idx];
}

// demb[b][c] = diffusion_emb[b] . dp_w[c] + dp_b[c]
__global__ void k_demb(const float* __restrict__ de, const float* __restrict__ dp_w,
                       const float* __restrict__ dp_b, float* __restrict__ demb)
{
  int c = threadIdx.x, b = blockIdx.x;
  float acc = dp_b[c];
  for (int d = 0; d < DEMBn; ++d) acc += de[b * DEMBn + d] * dp_w[c * DEMBn + d];
  demb[b * Cn + c] = acc;
}

// sa_in[b][n][l*C+c] = x[b][c][l*N+n] + demb[b][c] + cond_b[c] + pre_rows[(b,t=l*N+n)][c]
__global__ void k_build_sain(const float* __restrict__ x, const float* __restrict__ demb,
                             const float* __restrict__ cond_b, const float* __restrict__ pre,
                             float* __restrict__ sa_in)
{
  long long idx = (long long)blockIdx.x * blockDim.x + threadIdx.x;
  if (idx >= (long long)Bn * Nn * Ln * Cn) return;
  int c = (int)(idx & 63);
  int l = (int)((idx >> 6) % Ln);
  int n = (int)((idx / ((long long)Cn * Ln)) % Nn);
  int b = (int)(idx / ((long long)Cn * Ln * Nn));
  long long t = (long long)l * Nn + n;
  float v = x[((long long)b * Cn + c) * LNn + t] + demb[b * Cn + c] + cond_b[c]
          + pre[((long long)b * LNn + t) * Cn + c];
  sa_in[idx] = v;
}

// f1[b,n] = h[b,n,:] . a[:OUTC];  f2[b,n] = h[b,n,:] . a[OUTC:]
__global__ void __launch_bounds__(256)
k_f1f2(const float* __restrict__ h, const float* __restrict__ a,
       float* __restrict__ f1, float* __restrict__ f2)
{
  __shared__ float r1[256], r2[256];
  const float* hr = h + (long long)blockIdx.x * OUTCn;
  float s1 = 0.f, s2 = 0.f;
  for (int k = threadIdx.x; k < OUTCn; k += 256) {
    float hv = hr[k];
    s1 += hv * a[k];
    s2 += hv * a[OUTCn + k];
  }
  r1[threadIdx.x] = s1; r2[threadIdx.x] = s2;
  __syncthreads();
  for (int s = 128; s > 0; s >>= 1) {
    if (threadIdx.x < s) { r1[threadIdx.x] += r1[threadIdx.x + s];
                           r2[threadIdx.x] += r2[threadIdx.x + s]; }
    __syncthreads();
  }
  if (threadIdx.x == 0) { f1[blockIdx.x] = r1[0]; f2[blockIdx.x] = r2[0]; }
}

// adj = softmax(relu(GL@GL.T), axis=-1) + I   (one block per row i)
__global__ void __launch_bounds__(128)
k_adj(const float* __restrict__ GL, float* __restrict__ adj)
{
  __shared__ float red[128];
  int i = blockIdx.x, j = threadIdx.x;
  float e = 0.f;
  if (j < Nn) {
    float acc = 0.f;
    for (int d = 0; d < Gn; ++d) acc += GL[i * Gn + d] * GL[j * Gn + d];
    e = fmaxf(acc, 0.f);
  }
  red[j] = (j < Nn) ? e : -3.4e38f;
  __syncthreads();
  for (int s = 64; s > 0; s >>= 1) {
    if (j < s) red[j] = fmaxf(red[j], red[j + s]);
    __syncthreads();
  }
  float mx = red[0];
  __syncthreads();
  float ex = (j < Nn) ? __expf(e - mx) : 0.f;
  red[j] = ex;
  __syncthreads();
  for (int s = 64; s > 0; s >>= 1) {
    if (j < s) red[j] += red[j + s];
    __syncthreads();
  }
  if (j < Nn) adj[i * Nn + j] = ex / red[0] + (i == j ? 1.f : 0.f);
}

// att0[b,i,j] = adj[i,j] > 0 ? leaky_relu(f1[b,i]+f2[b,j], 0.01) : -1e12
__global__ void k_att0(const float* __restrict__ f1, const float* __restrict__ f2,
                       const float* __restrict__ adj, float* __restrict__ att0)
{
  int idx = blockIdx.x * blockDim.x + threadIdx.x;
  if (idx >= Bn * Nn * Nn) return;
  int j = idx % Nn, i = (idx / Nn) % Nn, b = idx / (Nn * Nn);
  float e = f1[b * Nn + i] + f2[b * Nn + j];
  e = (e > 0.f) ? e : 0.01f * e;
  att0[idx] = (adj[i * Nn + j] > 0.f) ? e : -1e12f;
}

// per (b,j): top-20 over i of att0[b,:,j] -> mask[b,i,j]
__global__ void k_topk(const float* __restrict__ att0, float* __restrict__ maskm)
{
  int id = blockIdx.x * blockDim.x + threadIdx.x;
  if (id >= Bn * Nn) return;
  int b = id / Nn, j = id % Nn;
  float col[Nn];
  for (int i = 0; i < Nn; ++i) {
    col[i] = att0[((long long)b * Nn + i) * Nn + j];
    maskm[((long long)b * Nn + i) * Nn + j] = 0.f;
  }
  for (int sel = 0; sel < TOPKn; ++sel) {
    int best = 0; float bv = col[0];
    for (int i = 1; i < Nn; ++i) if (col[i] > bv) { bv = col[i]; best = i; }
    maskm[((long long)b * Nn + best) * Nn + j] = 1.f;
    col[best] = -3.4e38f;
  }
}

// attS[b,i,:] = softmax_j(att0*mask)   (one block per (b,i))
__global__ void __launch_bounds__(128)
k_softmax_att(const float* __restrict__ att0, const float* __restrict__ maskm,
              float* __restrict__ attS)
{
  __shared__ float red[128];
  int bi = blockIdx.x, j = threadIdx.x;
  long long base = (long long)bi * Nn;
  float v = (j < Nn) ? att0[base + j] * maskm[base + j] : -3.4e38f;
  red[j] = v;
  __syncthreads();
  for (int s = 64; s > 0; s >>= 1) {
    if (j < s) red[j] = fmaxf(red[j], red[j + s]);
    __syncthreads();
  }
  float mx = red[0];
  __syncthreads();
  float ex = (j < Nn) ? __expf(v - mx) : 0.f;
  red[j] = ex;
  __syncthreads();
  for (int s = 64; s > 0; s >>= 1) {
    if (j < s) red[j] += red[j + s];
    __syncthreads();
  }
  if (j < Nn) attS[base + j] = ex / red[0];
}

// lstm_x[(b*N+n)*L*C + l*C + c] = hp[b][l][c*L + n]
__global__ void k_build_lstmx(const float* __restrict__ hp, float* __restrict__ lx)
{
  long long idx = (long long)blockIdx.x * blockDim.x + threadIdx.x;
  if (idx >= (long long)Bn * Nn * Ln * Cn) return;
  int c  = (int)(idx & 63);
  long long tmp = idx >> 6;
  int l2 = (int)(tmp % Ln);
  int bn = (int)(tmp / Ln);
  int b = bn / Nn, n2 = bn % Nn;
  lx[idx] = hp[((long long)b * Nn + l2) * OUTCn + c * Ln + n2];
}

// LSTM: T = B*N = 800 sequential steps, batch lanes = L = 100 (independent).
// 25 blocks x 4 lanes; thread (lane,g) computes 4 gates; h-state in LDS.
__global__ void __launch_bounds__(256)
k_lstm(const float* __restrict__ lx, const float* __restrict__ Wih,
       const float* __restrict__ Whh, const float* __restrict__ bih,
       const float* __restrict__ bhh, float* __restrict__ hsout)
{
  __shared__ float hsh[4][Gn];
  __shared__ float xsh[4][Cn];
  int tid = threadIdx.x, ty = tid >> 6, g = tid & 63;
  int l2 = blockIdx.x * 4 + ty;           // 0..99
  float cst = 0.f;
  hsh[ty][g] = 0.f;
  __syncthreads();
  const float bi0 = bih[g] + bhh[g];
  const float bf0 = bih[Gn + g] + bhh[Gn + g];
  const float bg0 = bih[2 * Gn + g] + bhh[2 * Gn + g];
  const float bo0 = bih[3 * Gn + g] + bhh[3 * Gn + g];
  for (int t = 0; t < Bn * Nn; ++t) {
    xsh[ty][g] = lx[((long long)t * Ln + l2) * Cn + g];
    __syncthreads();
    float gi = bi0, gf = bf0, gg = bg0, go = bo0;
    for (int c2 = 0; c2 < Cn; ++c2) {
      float xv = xsh[ty][c2], hv = hsh[ty][c2];
      gi += xv * Wih[g * Cn + c2]            + hv * Whh[g * Gn + c2];
      gf += xv * Wih[(Gn + g) * Cn + c2]     + hv * Whh[(Gn + g) * Gn + c2];
      gg += xv * Wih[(2 * Gn + g) * Cn + c2] + hv * Whh[(2 * Gn + g) * Gn + c2];
      go += xv * Wih[(3 * Gn + g) * Cn + c2] + hv * Whh[(3 * Gn + g) * Gn + c2];
    }
    cst = sigm(gf) * cst + sigm(gi) * tanhf(gg);
    float hnew = sigm(go) * tanhf(cst);
    __syncthreads();
    hsh[ty][g] = hnew;
    hsout[((long long)t * Ln + l2) * Gn + g] = hnew;
    __syncthreads();
  }
}

// X_tr[(n*800 + b*100 + l)*64 + c] = y2mat[(b*10000 + n*100 + l)*64 + c]
__global__ void k_build_xtr(const float* __restrict__ y2, float* __restrict__ xtr)
{
  long long idx = (long long)blockIdx.x * blockDim.x + threadIdx.x;
  if (idx >= (long long)Bn * LNn * Cn) return;
  int c = (int)(idx & 63);
  long long r = idx >> 6;
  int l = (int)(r % Ln);
  int b = (int)((r / Ln) % Bn);
  int n = (int)(r / (Ln * Bn));
  xtr[idx] = y2[((long long)b * LNn + (long long)n * Ln + l) * Cn + c];
}

// MHA core, hd=8, S=100: one block per (bl, head).
__global__ void __launch_bounds__(128)
k_mha(const float* __restrict__ qkv, float* __restrict__ ctx)
{
  __shared__ float qs[Nn][8], ks[Nn][8], vs[Nn][8];
  int bl = blockIdx.x >> 3, hh = blockIdx.x & 7;
  for (int idx = threadIdx.x; idx < Nn * 8; idx += 128) {
    int s = idx >> 3, d = idx & 7;
    long long base = ((long long)s * (Bn * Ln) + bl) * (3 * Cn) + hh * 8 + d;
    qs[s][d] = qkv[base];
    ks[s][d] = qkv[base + Cn];
    vs[s][d] = qkv[base + 2 * Cn];
  }
  __syncthreads();
  int s = threadIdx.x;
  if (s < Nn) {
    float sc[Nn];
    float mx = -3.4e38f;
    for (int t = 0; t < Nn; ++t) {
      float acc = 0.f;
      for (int d = 0; d < 8; ++d) acc += qs[s][d] * ks[t][d];
      acc *= 0.35355339059327373f;   // 1/sqrt(8)
      sc[t] = acc; mx = fmaxf(mx, acc);
    }
    float ssum = 0.f;
    for (int t = 0; t < Nn; ++t) { sc[t] = __expf(sc[t] - mx); ssum += sc[t]; }
    float inv = 1.f / ssum;
    float o[8] = {0.f,0.f,0.f,0.f,0.f,0.f,0.f,0.f};
    for (int t = 0; t < Nn; ++t) {
      float p = sc[t] * inv;
      for (int d = 0; d < 8; ++d) o[d] += p * vs[t][d];
    }
    for (int d = 0; d < 8; ++d)
      ctx[((long long)s * (Bn * Ln) + bl) * Cn + hh * 8 + d] = o[d];
  }
}

// LayerNorm over last dim 64, one block (64 threads) per row
__global__ void __launch_bounds__(64)
k_ln(const float* __restrict__ in, const float* __restrict__ g,
     const float* __restrict__ b, float* __restrict__ out)
{
  __shared__ float red[64];
  long long row = blockIdx.x;
  int c = threadIdx.x;
  float v = in[row * Cn + c];
  red[c] = v;
  __syncthreads();
  for (int s = 32; s > 0; s >>= 1) { if (c < s) red[c] += red[c + s]; __syncthreads(); }
  float mu = red[0] * (1.f / 64.f);
  __syncthreads();
  float dv = v - mu;
  red[c] = dv * dv;
  __syncthreads();
  for (int s = 32; s > 0; s >>= 1) { if (c < s) red[c] += red[c + s]; __syncthreads(); }
  float var = red[0] * (1.f / 64.f);
  out[row * Cn + c] = dv * rsqrtf(var + 1e-5f) * g[c] + b[c];
}

// gated activation: z = sigmoid(y5[:, :64]) * tanh(y5[:, 64:])
__global__ void k_gate(const float* __restrict__ y5, float* __restrict__ z)
{
  long long idx = (long long)blockIdx.x * blockDim.x + threadIdx.x;
  if (idx >= (long long)Bn * LNn * Cn) return;
  long long r = idx >> 6;
  int c = (int)(idx & 63);
  z[idx] = sigm(y5[r * 128 + c]) * tanhf(y5[r * 128 + Cn + c]);
}

// res = (x + residual)/sqrt(2); skip -> second half of d_out
__global__ void k_final(const float* __restrict__ x, const float* __restrict__ y6,
                        float* __restrict__ out)
{
  long long idx = (long long)blockIdx.x * blockDim.x + threadIdx.x;
  if (idx >= (long long)Bn * Cn * LNn) return;
  int u = (int)(idx % LNn);
  int c = (int)((idx / LNn) % Cn);
  int b = (int)(idx / ((long long)Cn * LNn));
  int n = u / Ln, l = u % Ln;
  long long r = (long long)n * (Bn * Ln) + b * Ln + l;
  out[idx] = (x[idx] + y6[r * 128 + c]) * 0.70710678118654752f;
  out[(long long)Bn * Cn * LNn + idx] = y6[r * 128 + Cn + c];
}

// ---------------------------------------------------------------------------
extern "C" void kernel_launch(void* const* d_in, const int* in_sizes, int n_in,
                              void* d_out, int out_size, void* d_ws, size_t ws_size,
                              hipStream_t stream)
{
  (void)in_sizes; (void)n_in; (void)out_size; (void)ws_size;
  const float* x       = (const float*)d_in[0];
  const float* ci      = (const float*)d_in[1];
  const float* de      = (const float*)d_in[2];
  const float* dp_w    = (const float*)d_in[3];
  const float* dp_b    = (const float*)d_in[4];
  const float* cond_w  = (const float*)d_in[5];
  const float* cond_b  = (const float*)d_in[6];
  const float* mid_w   = (const float*)d_in[7];
  const float* mid_b   = (const float*)d_in[8];
  const float* blk_ow  = (const float*)d_in[9];
  const float* blk_ob  = (const float*)d_in[10];
  const float* sa_W    = (const float*)d_in[11];
  const float* sa_a    = (const float*)d_in[12];
  const float* sa_GL   = (const float*)d_in[13];
  const float* Wih     = (const float*)d_in[14];
  const float* Whh     = (const float*)d_in[15];
  const float* bih     = (const float*)d_in[16];
  const float* bhh     = (const float*)d_in[17];
  const float* lin_w   = (const float*)d_in[18];
  const float* lin_b   = (const float*)d_in[19];
  const float* tin_w   = (const float*)d_in[20];
  const float* tin_b   = (const float*)d_in[21];
  const float* tow     = (const float*)d_in[22];
  const float* tob     = (const float*)d_in[23];
  const float* f1w     = (const float*)d_in[24];
  const float* f1b     = (const float*)d_in[25];
  const float* f2w     = (const float*)d_in[26];
  const float* f2b     = (const float*)d_in[27];
  const float* ln1g    = (const float*)d_in[28];
  const float* ln1b    = (const float*)d_in[29];
  const float* ln2g    = (const float*)d_in[30];
  const float* ln2b    = (const float*)d_in[31];
  float* out = (float*)d_out;
  float* ws  = (float*)d_ws;

  // static slot allocator (floats), with lifetime-based reuse
  const long long SLOT0 = 0;         // 15.36M f : ci_t / lstm_x / qkv / y5
  const long long SLOT1 = 15360000;  //  5.12M f : pre_rows / hsmat / ln1 / z
  const long long SLOT2 = 20480000;  //  5.12M f : sa_in / y2mat / ctx / ff1 / tout
  const long long SLOT3 = 25600000;  //  5.12M f : h / X_tr / ff2pre
  const long long SLOT4 = 30720000;  //  small   : demb,f1,f2,adj,att0,mask,attS
  const long long SLOT5 = 31000000;  // 10.24M f : hp / attnpre / y6

  float* ci_t    = ws + SLOT0;
  float* lstm_x  = ws + SLOT0;
  float* qkv     = ws + SLOT0;
  float* y5      = ws + SLOT0;
  float* pre     = ws + SLOT1;
  float* hsmat   = ws + SLOT1;
  float* ln1     = ws + SLOT1;
  float* zg      = ws + SLOT1;
  float* sa_in   = ws + SLOT2;
  float* y2mat   = ws + SLOT2;
  float* ctx     = ws + SLOT2;
  float* ff1     = ws + SLOT2;
  float* tout    = ws + SLOT2;
  float* hmat    = ws + SLOT3;
  float* xtr     = ws + SLOT3;
  float* ff2pre  = ws + SLOT3;
  float* demb    = ws + SLOT4;            // 512
  float* f1      = ws + SLOT4 + 1000;     // 800
  float* f2      = ws + SLOT4 + 2000;     // 800
  float* adj     = ws + SLOT4 + 4000;     // 10000
  float* att0    = ws + SLOT4 + 16000;    // 80000
  float* maskm   = ws + SLOT4 + 96000;    // 80000
  float* attS    = ws + SLOT4 + 176000;   // 80000
  float* hp      = ws + SLOT5;
  float* attnpre = ws + SLOT5;
  float* y6      = ws + SLOT5;

  auto gemm = [&](const float* A, long long lda, long long sA,
                  const float* Bw, long long ldb, long long sB, int transB,
                  float* C, long long ldc, long long sC,
                  const float* bias, const float* Add,
                  int M, int N, int K, int act, int batch) {
    dim3 g((unsigned)(N / 64), (unsigned)((M + 63) / 64), (unsigned)batch);
    gemm_wmma<<<g, 128, 0, stream>>>(A, lda, sA, Bw, ldb, sB, transB,
                                     C, ldc, sC, bias, Add, M, N, K, act);
  };
  auto blocks = [](long long n) { return (unsigned)((n + 255) / 256); };

  const long long EW = (long long)Bn * Cn * LNn;   // 5.12M

  // 1) ci -> ci_t  (position-major rows for X@W.T form)
  k_transpose_ci<<<blocks((long long)Bn * SIDEn * LNn), 256, 0, stream>>>(ci, ci_t);
  // 2) diffusion embedding projection
  k_demb<<<Bn, Cn, 0, stream>>>(de, dp_w, dp_b, demb);
  // 3) cond projection: pre_rows[80000x64] = ci_t @ cond_w.T   (K=144, guarded)
  gemm(ci_t, SIDEn, 0, cond_w, SIDEn, 0, 1, pre, Cn, 0,
       nullptr, nullptr, Bn * LNn, Cn, SIDEn, ACT_NONE, 1);
  // 4) assemble sparse-attention input [B][N][L*C]
  k_build_sain<<<blocks(EW), 256, 0, stream>>>(x, demb, cond_b, pre, sa_in);
  // 5) DOMINANT GEMM: h[b] = sa_in[b] @ sa_W   (800x6400x6400, sa_W L2-resident)
  gemm(sa_in, OUTCn, (long long)Nn * OUTCn, sa_W, OUTCn, 0, 0,
       hmat, OUTCn, (long long)Nn * OUTCn, nullptr, nullptr,
       Nn, OUTCn, OUTCn, ACT_NONE, Bn);
  // 6) GAT scores f1,f2
  k_f1f2<<<Bn * Nn, 256, 0, stream>>>(hmat, sa_a, f1, f2);
  // 7) adjacency softmax + I
  k_adj<<<Nn, 128, 0, stream>>>(sa_GL, adj);
  // 8) raw attention
  k_att0<<<blocks(Bn * Nn * Nn), 256, 0, stream>>>(f1, f2, adj, att0);
  // 9) column-wise top-20 mask
  k_topk<<<blocks(Bn * Nn), 256, 0, stream>>>(att0, maskm);
  // 10) row softmax of att*mask
  k_softmax_att<<<Bn * Nn, 128, 0, stream>>>(att0, maskm, attS);
  // 11) hp[b] = relu(attS[b] @ h[b])
  gemm(attS, Nn, (long long)Nn * Nn, hmat, OUTCn, (long long)Nn * OUTCn, 0,
       hp, OUTCn, (long long)Nn * OUTCn, nullptr, nullptr,
       Nn, OUTCn, Nn, ACT_RELU, Bn);
  // 12) LSTM input gather
  k_build_lstmx<<<blocks(EW), 256, 0, stream>>>(hp, lstm_x);
  // 13) sequential LSTM (800 steps, 100 independent lanes)
  k_lstm<<<25, 256, 0, stream>>>(lstm_x, Wih, Whh, bih, bhh, hsmat);
  // 14) y2 = silu(hs @ lin_w.T + lin_b)
  gemm(hsmat, Gn, 0, lin_w, Gn, 0, 1, y2mat, Cn, 0,
       lin_b, nullptr, Bn * LNn, Cn, Gn, ACT_SILU, 1);
  // 15) token-major rows for transformer
  k_build_xtr<<<blocks(EW), 256, 0, stream>>>(y2mat, xtr);
  // 16) qkv = X @ in_w.T + in_b
  gemm(xtr, Cn, 0, tin_w, Cn, 0, 1, qkv, 3 * Cn, 0,
       tin_b, nullptr, Bn * LNn, 3 * Cn, Cn, ACT_NONE, 1);
  // 17) attention core (hd=8)
  k_mha<<<Bn * Ln * 8, 128, 0, stream>>>(qkv, ctx);
  // 18) out-proj + residual
  gemm(ctx, Cn, 0, tow, Cn, 0, 1, attnpre, Cn, 0,
       tob, xtr, Bn * LNn, Cn, Cn, ACT_NONE, 1);
  // 19) layernorm 1
  k_ln<<<Bn * LNn, 64, 0, stream>>>(attnpre, ln1g, ln1b, ln1);
  // 20) FFN up + exact gelu
  gemm(ln1, Cn, 0, f1w, Cn, 0, 1, ff1, Cn, 0,
       f1b, nullptr, Bn * LNn, Cn, Cn, ACT_GELU, 1);
  // 21) FFN down + residual
  gemm(ff1, Cn, 0, f2w, Cn, 0, 1, ff2pre, Cn, 0,
       f2b, ln1, Bn * LNn, Cn, Cn, ACT_NONE, 1);
  // 22) layernorm 2
  k_ln<<<Bn * LNn, 64, 0, stream>>>(ff2pre, ln2g, ln2b, tout);
  // 23) mid projection (128 out)
  gemm(tout, Cn, 0, mid_w, Cn, 0, 1, y5, 2 * Cn, 0,
       mid_b, nullptr, Bn * LNn, 2 * Cn, Cn, ACT_NONE, 1);
  // 24) gated activation
  k_gate<<<blocks(EW), 256, 0, stream>>>(y5, zg);
  // 25) output projection (128 out)
  gemm(zg, Cn, 0, blk_ow, Cn, 0, 1, y6, 2 * Cn, 0,
       blk_ob, nullptr, Bn * LNn, 2 * Cn, Cn, ACT_NONE, 1);
  // 26) residual/skip split -> d_out
  k_final<<<blocks(EW), 256, 0, stream>>>(x, y6, out);
}
